// MultiHeadAttention_36386962932347
// MI455X (gfx1250) — matrix operand order repaired and checked by reading, not compiled, not run on hardware
//
#include <hip/hip_runtime.h>

typedef unsigned short u16;
typedef unsigned int   u32;

typedef __attribute__((ext_vector_type(16))) __bf16 v16bf;
typedef __attribute__((ext_vector_type(8)))  float  v8f;
typedef __attribute__((ext_vector_type(4)))  int    v4i;

union AF { v16bf v; u32 u[8]; };
union CF { v8f v; float f[8]; };

// Detect the gfx1250 async global->LDS builtin (ASYNCcnt path, ISA §08).
#if defined(__HIP_DEVICE_COMPILE__)
#  if __has_builtin(__builtin_amdgcn_global_load_async_to_lds_b128)
#    define USE_ASYNC_LDS 1
#  else
#    define USE_ASYNC_LDS 0
#    pragma message("gfx1250: global_load_async_to_lds builtin NOT available -> sync staging fallback")
#  endif
#else
#  define USE_ASYNC_LDS 0
#endif

#if USE_ASYNC_LDS
typedef __attribute__((address_space(1))) v4i GV4;   // global int4*
typedef __attribute__((address_space(3))) v4i LV4;   // LDS int4*
#endif

// Problem constants (match reference)
constexpr int kB  = 4;
constexpr int kS  = 2048;
constexpr int kD  = 1024;
constexpr int kH  = 16;
constexpr int kHD = 64;
constexpr int kFF = 4096;
constexpr int kBS = kB * kS;          // 8192 rows

__device__ __forceinline__ u16 f2bf(float x) {
    union { float f; u32 u; } c; c.f = x;
    u32 u = c.u;
    u = (u + 0x7FFFu + ((u >> 16) & 1u)) >> 16;   // round-to-nearest-even
    return (u16)u;
}

__device__ __forceinline__ v8f wmma_bf16(const AF& a, const AF& b, v8f c) {
    return __builtin_amdgcn_wmma_f32_16x16x32_bf16(
        /*neg_a=*/false, a.v, /*neg_b=*/false, b.v,
        /*c_mod=*/(short)0, c, /*reuse_a=*/false, /*reuse_b=*/false);
}

// 16-byte global -> LDS copy (async when available)
__device__ __forceinline__ void cp16(u16* lds, const u16* g) {
#if USE_ASYNC_LDS
    __builtin_amdgcn_global_load_async_to_lds_b128((GV4*)g, (LV4*)lds, 0, 0);
#else
    *(uint4*)lds = *(const uint4*)g;
#endif
}
__device__ __forceinline__ void wait_async_pending5() {
#if USE_ASYNC_LDS
#  if __has_builtin(__builtin_amdgcn_s_wait_asynccnt)
    __builtin_amdgcn_s_wait_asynccnt(5);
#  else
    asm volatile("s_wait_asynccnt 5" ::: "memory");
#  endif
#endif
}
__device__ __forceinline__ void wait_async_all() {
#if USE_ASYNC_LDS
#  if __has_builtin(__builtin_amdgcn_s_wait_asynccnt)
    __builtin_amdgcn_s_wait_asynccnt(0);
#  else
    asm volatile("s_wait_asynccnt 0" ::: "memory");
#  endif
#endif
}

// ------------------------------------------- fp32 [K,N] -> bf16 [N,K] (tiled)
__global__ __launch_bounds__(256) void convT_bf16_kernel(const float* __restrict__ in,
                                                         u16* __restrict__ out,
                                                         int Kd, int Nd) {
    __shared__ u16 tile[32][33];
    const int tx = threadIdx.x & 31;
    const int ty = threadIdx.x >> 5;          // 0..7
    const int n = blockIdx.x * 32 + tx;
    for (int i = 0; i < 32; i += 8) {
        int k = blockIdx.y * 32 + ty + i;
        tile[ty + i][tx] = f2bf(in[(size_t)k * Nd + n]);
    }
    __syncthreads();
    const int k2 = blockIdx.y * 32 + tx;
    for (int i = 0; i < 32; i += 8) {
        int n2 = blockIdx.x * 32 + ty + i;
        out[(size_t)n2 * Kd + k2] = tile[tx][ty + i];
    }
}

// ---------------------------------------------------------------- LayerNorm #1
__global__ __launch_bounds__(256) void ln1_kernel(const float* __restrict__ x,
                                                  const float* __restrict__ g,
                                                  const float* __restrict__ bta,
                                                  u16* __restrict__ hbf) {
    const int row = blockIdx.x;
    const float* xr = x + (size_t)row * kD;
    float vals[4], s = 0.f, s2 = 0.f;
    for (int i = 0; i < 4; ++i) {
        float t = xr[threadIdx.x + 256 * i];
        vals[i] = t; s += t; s2 += t * t;
    }
    for (int off = 1; off < 32; off <<= 1) {
        s  += __shfl_xor(s,  off, 32);
        s2 += __shfl_xor(s2, off, 32);
    }
    __shared__ float ss[8], ss2[8];
    int w = threadIdx.x >> 5, lane = threadIdx.x & 31;
    if (lane == 0) { ss[w] = s; ss2[w] = s2; }
    __syncthreads();
    if (w == 0) {
        s  = (lane < 8) ? ss[lane]  : 0.f;
        s2 = (lane < 8) ? ss2[lane] : 0.f;
        for (int off = 1; off < 8; off <<= 1) {
            s += __shfl_xor(s, off, 32); s2 += __shfl_xor(s2, off, 32);
        }
        if (lane == 0) { ss[0] = s; ss2[0] = s2; }
    }
    __syncthreads();
    float mu = ss[0] * (1.f / kD);
    float rs = rsqrtf(ss2[0] * (1.f / kD) - mu * mu + 1e-5f);
    for (int i = 0; i < 4; ++i) {
        int c = threadIdx.x + 256 * i;
        hbf[(size_t)row * kD + c] = f2bf((vals[i] - mu) * rs * g[c] + bta[c]);
    }
}

// ----------------------------------------- LayerNorm #2 on (x + attn) residual
__global__ __launch_bounds__(256) void ln2_kernel(const float* __restrict__ x,
                                                  const float* __restrict__ attn,
                                                  const float* __restrict__ g,
                                                  const float* __restrict__ bta,
                                                  float* __restrict__ fout,
                                                  u16* __restrict__ fbf) {
    const int row = blockIdx.x;
    const float* xr = x    + (size_t)row * kD;
    const float* ar = attn + (size_t)row * kD;
    float vals[4], s = 0.f, s2 = 0.f;
    for (int i = 0; i < 4; ++i) {
        int c = threadIdx.x + 256 * i;
        float t = xr[c] + ar[c];
        vals[i] = t; s += t; s2 += t * t;
    }
    for (int off = 1; off < 32; off <<= 1) {
        s += __shfl_xor(s, off, 32); s2 += __shfl_xor(s2, off, 32);
    }
    __shared__ float ss[8], ss2[8];
    int w = threadIdx.x >> 5, lane = threadIdx.x & 31;
    if (lane == 0) { ss[w] = s; ss2[w] = s2; }
    __syncthreads();
    if (w == 0) {
        s  = (lane < 8) ? ss[lane]  : 0.f;
        s2 = (lane < 8) ? ss2[lane] : 0.f;
        for (int off = 1; off < 8; off <<= 1) {
            s += __shfl_xor(s, off, 32); s2 += __shfl_xor(s2, off, 32);
        }
        if (lane == 0) { ss[0] = s; ss2[0] = s2; }
    }
    __syncthreads();
    float mu = ss[0] * (1.f / kD);
    float rs = rsqrtf(ss2[0] * (1.f / kD) - mu * mu + 1e-5f);
    for (int i = 0; i < 4; ++i) {
        int c = threadIdx.x + 256 * i;
        float fv = (vals[i] - mu) * rs * g[c] + bta[c];
        fout[(size_t)row * kD + c] = fv;
        fbf [(size_t)row * kD + c] = f2bf(fv);
    }
}

// ---------------------------------------------------------------- bf16 GEMM
// C[M,N] = A[M,K] * B[K,N] + bias (+ReLU) -> bf16   (MODE 0)
//        = ... + bias + res[M,N]          -> fp32   (MODE 1)
// B is supplied TRANSPOSED in global memory: Bt[N,K] (bf16), so both tiles are
// contiguous 16B copies -> async global->LDS, double-buffered pipeline.
// Block 256 thr (8 waves); tile 256(M) x 64(N) x 32(K); wave tile 32x64.
template <int MODE, bool RELU>
__global__ __launch_bounds__(256) void gemm_bf16_k(const u16* __restrict__ A,
                                                   const u16* __restrict__ Bt,
                                                   const float* __restrict__ bias,
                                                   void* __restrict__ outp,
                                                   const float* __restrict__ res,
                                                   int M, int N, int K) {
    __shared__ u16 As[2][256 * 32];   // [buf][row][k]
    __shared__ u16 Bs[2][64 * 32];    // [buf][col][k]  (k-pairs contiguous)
    const int m0 = blockIdx.y * 256;
    const int n0 = blockIdx.x * 64;
    const int t = threadIdx.x;
    const int w = t >> 5, lane = t & 31;
    const int arow = lane & 15, half = lane >> 4, col = lane & 15;

    // staging coordinates: 5 x 16B per thread per K-step
    const int sr  = t >> 1;                 // A rows sr and sr+128
    const int scg = (t & 1) * 16;           // A col group
    const int bn  = t >> 2;                 // B row (=output col) 0..63
    const int bc  = (t & 3) * 8;            // B k group
    const u16* a_src0 = A  + (size_t)(m0 + sr) * K + scg;
    const u16* a_src1 = a_src0 + (size_t)128 * K;
    const u16* b_src  = Bt + (size_t)(n0 + bn) * K + bc;

    CF acc[2][4];
    for (int r = 0; r < 2; ++r)
        for (int f = 0; f < 4; ++f)
            for (int v = 0; v < 8; ++v) acc[r][f].f[v] = 0.f;

    const int nIter = K >> 5;
    auto stage = [&](int it, int buf) {
        const int k0 = it << 5;
        cp16(&As[buf][sr * 32 + scg],              a_src0 + k0);
        cp16(&As[buf][sr * 32 + scg + 8],          a_src0 + k0 + 8);
        cp16(&As[buf][(sr + 128) * 32 + scg],      a_src1 + k0);
        cp16(&As[buf][(sr + 128) * 32 + scg + 8],  a_src1 + k0 + 8);
        cp16(&Bs[buf][bn * 32 + bc],               b_src + k0);
    };

    stage(0, 0);
    for (int it = 0; it < nIter; ++it) {
        if (it + 1 < nIter) { stage(it + 1, (it + 1) & 1); wait_async_pending5(); }
        else                { wait_async_all(); }
        __syncthreads();                         // tile `it` visible to all waves
        const int buf = it & 1;
        AF a[2];
        for (int r = 0; r < 2; ++r)
            for (int v = 0; v < 8; ++v) {
                int kin = (v < 4) ? (half * 8 + 2 * v) : (16 + half * 8 + 2 * (v - 4));
                a[r].u[v] = *(const u32*)&As[buf][(w * 32 + r * 16 + arow) * 32 + kin];
            }
        for (int f = 0; f < 4; ++f) {
            AF b;
            for (int j = 0; j < 8; ++j)
                b.u[j] = *(const u32*)&Bs[buf][(f * 16 + col) * 32 + ((j < 4) ? (half * 8 + 2 * j) : (16 + half * 8 + 2 * (j - 4)))];
            acc[0][f].v = wmma_bf16(a[0], b, acc[0][f].v);
            acc[1][f].v = wmma_bf16(a[1], b, acc[1][f].v);
        }
        __syncthreads();                         // this buf free for reuse next+1
    }

    for (int r = 0; r < 2; ++r)
        for (int f = 0; f < 4; ++f) {
            int gcol = n0 + f * 16 + col;
            float bv = bias[gcol];
            for (int v = 0; v < 8; ++v) {
                int grow = m0 + w * 32 + r * 16 + half * 8 + v;
                float val = acc[r][f].f[v] + bv;
                if (RELU) val = fmaxf(val, 0.f);
                size_t idx = (size_t)grow * N + gcol;
                if (MODE == 0) ((u16*)outp)[idx] = f2bf(val);
                else           ((float*)outp)[idx] = val + res[idx];
            }
        }
}

// ---------------------------------------------------------------- attention
// Flash-attention, per (b, h): block = 128 thr (4 waves), each wave one 16-row
// Q tile.  S^T = K * Q^T keeps softmax stats per-lane (col = q); O^T = V^T * P^T
// lets the same per-lane stats rescale the accumulator directly.
__global__ __launch_bounds__(128) void attn_kernel(const u16* __restrict__ Q,
                                                   const u16* __restrict__ Kk,
                                                   const u16* __restrict__ Vv,
                                                   float* __restrict__ O) {
    __shared__ u16 VT[64 * 32];       // V tile transposed: [d][kv], block-shared
    __shared__ u16 PT[4][16 * 32];    // per-wave probs: [q][kv]
    const int w = threadIdx.x >> 5;
    const int lane = threadIdx.x & 31;
    const int b = blockIdx.z, h = blockIdx.y;
    const int q0 = (blockIdx.x * 4 + w) * 16;
    const size_t base = (size_t)b * kS * kD + (size_t)h * kHD;
    const int qc = lane & 15, half = lane >> 4;

    AF qb[2];   // Q^T as WMMA-B fragments (loop invariant)
    for (int f = 0; f < 2; ++f)
        for (int j = 0; j < 8; ++j) {
            int hd = f * 32 + ((j < 4) ? (half * 8 + 2 * j) : (16 + half * 8 + 2 * (j - 4)));
            qb[f].u[j] = *(const u32*)(Q + base + (size_t)(q0 + qc) * kD + hd);
        }

    float m = -1e30f, l = 0.f;
    CF o[4];
    for (int f = 0; f < 4; ++f)
        for (int v = 0; v < 8; ++v) o[f].f[v] = 0.f;

    constexpr float kScale = 0.125f * 1.4426950408889634f;   // 1/sqrt(64)*log2(e)

    for (int kv0 = 0; kv0 < kS; kv0 += 32) {
        __syncthreads();
        {   // stage V tile transposed
            int d0 = 2 * (threadIdx.x & 31);
            int kq = (threadIdx.x >> 5) * 8;
            for (int kv = kq; kv < kq + 8; kv += 2) {
                u32 a = *(const u32*)(Vv + base + (size_t)(kv0 + kv)     * kD + d0);
                u32 c = *(const u32*)(Vv + base + (size_t)(kv0 + kv + 1) * kD + d0);
                *(u32*)&VT[d0 * 32 + kv]       = (a & 0xFFFFu) | (c << 16);
                *(u32*)&VT[(d0 + 1) * 32 + kv] = (a >> 16) | (c & 0xFFFF0000u);
            }
        }

        CF st[2];
        for (int tt = 0; tt < 2; ++tt) {
            v8f acc = {};
            for (int f = 0; f < 2; ++f) {
                AF ka;
                for (int v = 0; v < 8; ++v) {
                    int kin = (v < 4) ? (half * 8 + 2 * v) : (16 + half * 8 + 2 * (v - 4));
                    ka.u[v] = *(const u32*)(Kk + base +
                              (size_t)(kv0 + tt * 16 + qc) * kD + f * 32 + kin);
                }
                acc = wmma_bf16(ka, qb[f], acc);
            }
            st[tt].v = acc;
            for (int v = 0; v < 8; ++v) st[tt].f[v] *= kScale;
        }
        __syncthreads();

        float tm = -1e30f;
        for (int v = 0; v < 8; ++v) tm = fmaxf(tm, fmaxf(st[0].f[v], st[1].f[v]));
        tm = fmaxf(tm, __shfl_xor(tm, 16, 32));
        float mn = fmaxf(m, tm);
        float alpha = exp2f(m - mn);
        float rsum = 0.f;
        for (int tt = 0; tt < 2; ++tt)
            for (int v = 0; v < 8; ++v) {
                float p = exp2f(st[tt].f[v] - mn);
                st[tt].f[v] = p;
                rsum += p;
            }
        rsum += __shfl_xor(rsum, 16, 32);
        l = l * alpha + rsum;
        m = mn;

        u16* pw = PT[w];
        for (int tt = 0; tt < 2; ++tt)
            for (int v = 0; v < 8; v += 2) {
                int kv = tt * 16 + half * 8 + v;
                u32 pk = (u32)f2bf(st[tt].f[v]) | ((u32)f2bf(st[tt].f[v + 1]) << 16);
                *(u32*)&pw[qc * 32 + kv] = pk;
            }
        asm volatile("s_wait_dscnt 0" ::: "memory");
        AF pb;
        for (int j = 0; j < 8; ++j)
            pb.u[j] = *(const u32*)&pw[qc * 32 + ((j < 4) ? (half * 8 + 2 * j) : (16 + half * 8 + 2 * (j - 4)))];

        for (int f = 0; f < 4; ++f)
            for (int v = 0; v < 8; ++v) o[f].f[v] *= alpha;
        for (int f = 0; f < 4; ++f) {
            AF va;
            for (int v = 0; v < 8; ++v) {
                int kin = (v < 4) ? (half * 8 + 2 * v) : (16 + half * 8 + 2 * (v - 4));
                va.u[v] = *(const u32*)&VT[(f * 16 + qc) * 32 + kin];
            }
            o[f].v = wmma_bf16(va, pb, o[f].v);
        }
    }

    float inv = 1.f / l;
    for (int f = 0; f < 4; ++f)
        for (int v = 0; v < 8; ++v) {
            int d = f * 16 + half * 8 + v;
            O[((size_t)b * kS + q0 + qc) * kD + (size_t)h * kHD + d] = o[f].f[v] * inv;
        }
}

// ---------------------------------------------------------------- launcher
extern "C" void kernel_launch(void* const* d_in, const int* in_sizes, int n_in,
                              void* d_out, int out_size, void* d_ws, size_t ws_size,
                              hipStream_t stream) {
    const float* x   = (const float*)d_in[0];
    const float* Wq  = (const float*)d_in[1];
    const float* bq  = (const float*)d_in[2];
    const float* Wk  = (const float*)d_in[3];
    const float* bk  = (const float*)d_in[4];
    const float* Wv  = (const float*)d_in[5];
    const float* bv  = (const float*)d_in[6];
    const float* g1  = (const float*)d_in[7];
    const float* be1 = (const float*)d_in[8];
    const float* g2  = (const float*)d_in[9];
    const float* be2 = (const float*)d_in[10];
    const float* W1  = (const float*)d_in[11];
    const float* b1  = (const float*)d_in[12];
    const float* W2  = (const float*)d_in[13];
    const float* b2  = (const float*)d_in[14];
    float* out = (float*)d_out;

    char* ws = (char*)d_ws;
    size_t off = 0;
    auto alloc = [&](size_t bytes) {
        void* p = ws + off;
        off += (bytes + 255) & ~(size_t)255;
        return p;
    };
    u16* wq_t = (u16*)alloc((size_t)kD * kD * 2);    // [D,D]   (N,K)
    u16* wk_t = (u16*)alloc((size_t)kD * kD * 2);
    u16* wv_t = (u16*)alloc((size_t)kD * kD * 2);
    u16* w1_t = (u16*)alloc((size_t)kFF * kD * 2);   // [FF,D]
    u16* w2_t = (u16*)alloc((size_t)kD * kFF * 2);   // [D,FF]
    u16* h_bf = (u16*)alloc((size_t)kBS * kD * 2);
    u16* q_bf = (u16*)alloc((size_t)kBS * kD * 2);
    u16* k_bf = (u16*)alloc((size_t)kBS * kD * 2);
    u16* v_bf = (u16*)alloc((size_t)kBS * kD * 2);
    float* attn = (float*)alloc((size_t)kBS * kD * 4);
    float* f32r = (float*)alloc((size_t)kBS * kD * 4);
    u16* f_bf = (u16*)alloc((size_t)kBS * kD * 2);
    u16* u_bf = (u16*)alloc((size_t)kBS * kFF * 2);

    // 1) weights -> bf16, transposed to [N,K] for contiguous B-tile staging
    {
        dim3 gdd(kD / 32, kD / 32);
        convT_bf16_kernel<<<gdd, 256, 0, stream>>>(Wq, wq_t, kD, kD);
        convT_bf16_kernel<<<gdd, 256, 0, stream>>>(Wk, wk_t, kD, kD);
        convT_bf16_kernel<<<gdd, 256, 0, stream>>>(Wv, wv_t, kD, kD);
        dim3 g1g(kFF / 32, kD / 32);   // W1 [D,FF] -> [FF,D]
        convT_bf16_kernel<<<g1g, 256, 0, stream>>>(W1, w1_t, kD, kFF);
        dim3 g2g(kD / 32, kFF / 32);   // W2 [FF,D] -> [D,FF]
        convT_bf16_kernel<<<g2g, 256, 0, stream>>>(W2, w2_t, kFF, kD);
    }

    // 2) LN1
    ln1_kernel<<<kBS, 256, 0, stream>>>(x, g1, be1, h_bf);

    // 3) Q/K/V projections
    {
        dim3 grid(kD / 64, kBS / 256);
        gemm_bf16_k<0, false><<<grid, 256, 0, stream>>>(h_bf, wq_t, bq, q_bf, nullptr, kBS, kD, kD);
        gemm_bf16_k<0, false><<<grid, 256, 0, stream>>>(h_bf, wk_t, bk, k_bf, nullptr, kBS, kD, kD);
        gemm_bf16_k<0, false><<<grid, 256, 0, stream>>>(h_bf, wv_t, bv, v_bf, nullptr, kBS, kD, kD);
    }

    // 4) attention
    {
        dim3 grid(kS / 64, kH, kB);
        attn_kernel<<<grid, 128, 0, stream>>>(q_bf, k_bf, v_bf, attn);
    }

    // 5) LN2 on (x + attn)
    ln2_kernel<<<kBS, 256, 0, stream>>>(x, attn, g2, be2, f32r, f_bf);

    // 6) FFN1: relu(f @ W1 + b1) -> bf16
    {
        dim3 grid(kFF / 64, kBS / 256);
        gemm_bf16_k<0, true><<<grid, 256, 0, stream>>>(f_bf, w1_t, b1, u_bf, nullptr, kBS, kFF, kD);
    }

    // 7) FFN2: out = f + (u @ W2 + b2) -> fp32
    {
        dim3 grid(kD / 64, kBS / 256);
        gemm_bf16_k<1, false><<<grid, 256, 0, stream>>>(u_bf, w2_t, b2, out, f32r, kBS, kD, kFF);
    }

    (void)in_sizes; (void)n_in; (void)out_size; (void)ws_size;
}
